// Head_730144441075
// MI455X (gfx1250) — compile-verified, matrix operand order checked by reading, not run on hardware
//
#include <hip/hip_runtime.h>
#include <hip/hip_bf16.h>
#include <stdint.h>

typedef __attribute__((ext_vector_type(16))) __bf16 v16bf;
typedef __attribute__((ext_vector_type(8)))  __bf16 v8bf;
typedef __attribute__((ext_vector_type(8)))  float  v8f;
typedef __attribute__((ext_vector_type(4)))  unsigned int u32x4;
typedef __attribute__((ext_vector_type(8)))  int i32x8;
typedef __attribute__((ext_vector_type(4)))  int i32x4;

#define BATCH 16
#define SEQ   2048
#define CIN   1024
#define HDIM  128

// LDS pitches (bf16 elements). 40 -> 80B row stride, 136 -> 272B row stride;
// both give conflict-free 16B fragment reads across the 16-lane groups, and both
// are producible by the TDM's pad_interval/pad_amount LDS padding.
#define PITCH_A   40
#define PITCH_KV  136
#define PITCH_P   40

__device__ inline v8f zero8() {
    v8f z;
#pragma unroll
    for (int i = 0; i < 8; ++i) z[i] = 0.0f;
    return z;
}

// 32-byte fragment load as two 16B chunks (LDS rows here are only 16B aligned).
__device__ inline v16bf load16bf(const __bf16* p) {
    v16bf r;
    ((v8bf*)&r)[0] = *(const v8bf*)(p);
    ((v8bf*)&r)[1] = *(const v8bf*)(p + 8);
    return r;
}

__device__ inline v8f wmma_bf16(v16bf a, v16bf b, v8f c) {
    return __builtin_amdgcn_wmma_f32_16x16x32_bf16(
        /*neg_a=*/false, a, /*neg_b=*/false, b,
        /*c_mod=*/(short)0, c, /*reuse_a=*/false, /*reuse_b=*/false);
}

// ---------------------------------------------------------------------------
// TDM: async DMA of a 2D bf16 tile (tile_d0 x tile_d1 elements) from global
// memory into LDS, with per-row LDS padding to realize a banked pitch.
//   pad_interval code c: pad inserted every 2^(c+1) DWORDs stored
//   pad_amount  code a: pad of (a+1) DWORDs
// Tracked by TENSORcnt (S_WAIT_TENSORCNT).
// ---------------------------------------------------------------------------
__device__ inline void tdm_load_2d(uint32_t lds_off, const void* gptr,
                                   uint32_t tensor_d0, uint32_t tile_d0,
                                   uint32_t tile_d1, uint64_t stride_d0,
                                   uint32_t pad_interval, uint32_t pad_amount)
{
    const uint64_t ga  = (uint64_t)(uintptr_t)gptr;
    const uint32_t td1 = 1u << 20;              // generous dim1 (OOB headroom)

    u32x4 g0;
    g0[0] = 1u;                                  // count=1, user descriptor
    g0[1] = lds_off;                             // lds_addr [63:32]
    g0[2] = (uint32_t)ga;                        // global_addr low 32
    g0[3] = (uint32_t)((ga >> 32) & 0x01FFFFFFu) // global_addr [56:32]
          | (2u << 30);                          // type = 2 ("image")

    i32x8 g1;
    g1[0] = (int)((1u << 16)                     // data_size = 1 -> 2 bytes
                | (1u << 20)                     // pad_enable
                | (pad_interval << 22)
                | (pad_amount << 25));
    g1[1] = (int)((tensor_d0 & 0xFFFFu) << 16);  // tensor_dim0 [79:48] low half
    g1[2] = (int)(((tensor_d0 >> 16) & 0xFFFFu)
                | ((td1 & 0xFFFFu) << 16));      // dim0 hi | tensor_dim1 lo
    g1[3] = (int)(((td1 >> 16) & 0xFFFFu)
                | (tile_d0 << 16));              // tensor_dim1 hi | tile_dim0
    g1[4] = (int)(tile_d1 & 0xFFFFu);            // tile_dim1 (tile_dim2 = 0)
    g1[5] = (int)(uint32_t)(stride_d0 & 0xFFFFFFFFu);   // tensor_dim0_stride lo
    g1[6] = (int)(uint32_t)((stride_d0 >> 32) & 0xFFFFu);
    g1[7] = 0;

    const i32x4 z4 = {0, 0, 0, 0};
#if defined(__clang_major__) && (__clang_major__ >= 23)
    const i32x8 z8 = {0, 0, 0, 0, 0, 0, 0, 0};
    __builtin_amdgcn_tensor_load_to_lds(g0, g1, z4, z4, z8, 0);
#else
    __builtin_amdgcn_tensor_load_to_lds(g0, g1, z4, z4, 0);
#endif
}

// ---------------------------------------------------------------------------
// Kernel 1: fused QKV projection.  out = x @ W, x:[B*T,C] fp32, W:[C,H] fp32.
// blockIdx.z selects {K,Q,V}. Output bf16: K,Q as [B,T,H]; V transposed [B,H,T].
// (fp32 -> bf16 conversion happens in the staging, so no TDM here.)
// ---------------------------------------------------------------------------
__global__ __launch_bounds__(256) void qkv_proj_kernel(
    const float* __restrict__ x,
    const float* __restrict__ Wk,
    const float* __restrict__ Wq,
    const float* __restrict__ Wv,
    __bf16* __restrict__ outK,
    __bf16* __restrict__ outQ,
    __bf16* __restrict__ outVT)
{
    __shared__ __bf16 aTile[128 * PITCH_A];   // x tile   [row][k]
    __shared__ __bf16 wTile[128 * PITCH_A];   // W tile transposed [n][k]

    const int z = blockIdx.z;
    const float* __restrict__ W = (z == 0) ? Wk : (z == 1) ? Wq : Wv;
    __bf16* __restrict__ out    = (z == 0) ? outK : (z == 1) ? outQ : outVT;
    const bool transpose_out = (z == 2);

    const int tid  = threadIdx.x;
    const int lane = tid & 31;
    const int wave = tid >> 5;
    const int rowBase = blockIdx.x * 128;

    v8f acc[8];
#pragma unroll
    for (int i = 0; i < 8; ++i) acc[i] = zero8();

    for (int k0 = 0; k0 < CIN; k0 += 32) {
        __syncthreads();
        // Stage A: 128 rows x 32 k, fp32 -> bf16 (1024 float4 chunks, 4/thread)
        for (int c = tid; c < 1024; c += 256) {
            const int r  = c >> 3;
            const int kk = (c & 7) * 4;
            const float4 f = *(const float4*)(x + (size_t)(rowBase + r) * CIN + k0 + kk);
            __bf16* dst = &aTile[r * PITCH_A + kk];
            dst[0] = (__bf16)f.x; dst[1] = (__bf16)f.y;
            dst[2] = (__bf16)f.z; dst[3] = (__bf16)f.w;
        }
        // Stage W transposed: read [32 k][128 n] row-major, write [n][k]
        for (int c = tid; c < 1024; c += 256) {
            const int kk = c >> 5;
            const int nn = (c & 31) * 4;
            const float4 f = *(const float4*)(W + (size_t)(k0 + kk) * HDIM + nn);
            wTile[(nn + 0) * PITCH_A + kk] = (__bf16)f.x;
            wTile[(nn + 1) * PITCH_A + kk] = (__bf16)f.y;
            wTile[(nn + 2) * PITCH_A + kk] = (__bf16)f.z;
            wTile[(nn + 3) * PITCH_A + kk] = (__bf16)f.w;
        }
        __syncthreads();

        // A fragment: 16x32 bf16, lane<16 -> M=lane,k=0..15; lane>=16 -> k=16..31
        const int frow = wave * 16 + (lane & 15);
        const int fk   = (lane >> 4) * 16;
        const v16bf a = load16bf(&aTile[frow * PITCH_A + fk]);
#pragma unroll
        for (int ns = 0; ns < 8; ++ns) {
            const v16bf b = load16bf(&wTile[(ns * 16 + (lane & 15)) * PITCH_A + fk]);
            acc[ns] = wmma_bf16(a, b, acc[ns]);
        }
    }

    // Epilogue: C layout -> lane gives N (within 16), VGPR r gives M (r or r+8)
    const int rowInTile = wave * 16 + ((lane >> 4) * 8);
    const int n0 = lane & 15;
#pragma unroll
    for (int ns = 0; ns < 8; ++ns) {
        const int h = ns * 16 + n0;
#pragma unroll
        for (int r = 0; r < 8; ++r) {
            const int row = rowBase + rowInTile + r;
            const __bf16 val = (__bf16)acc[ns][r];
            if (!transpose_out) {
                out[(size_t)row * HDIM + h] = val;
            } else {
                const int bb = row >> 11;        // /SEQ
                const int tt = row & (SEQ - 1);
                out[((size_t)bb * HDIM + h) * SEQ + tt] = val;
            }
        }
    }
}

// ---------------------------------------------------------------------------
// Kernel 2: causal flash attention. q,k:[B,T,H] bf16, vT:[B,H,T] bf16.
// Block = 256 threads (8 waves), one batch x 128-query tile per block.
// K/V tiles are DMA'd into double-buffered LDS by the Tensor Data Mover
// (wave 0 issues, TENSORcnt + workgroup barrier synchronize), prefetching
// block kb+1 while all 8 waves run 16 WMMAs + online softmax on block kb.
// ---------------------------------------------------------------------------
__global__ __launch_bounds__(256) void attn_kernel(
    const __bf16* __restrict__ q,
    const __bf16* __restrict__ kmat,
    const __bf16* __restrict__ vT,
    float* __restrict__ out)
{
    __shared__ __bf16 kTile[2][32 * PITCH_KV];     // [buf][key][h]
    __shared__ __bf16 vTile[2][128 * PITCH_P];     // [buf][h][key]
    __shared__ __bf16 pTile[8][16 * PITCH_P];      // per-wave P bounce [q][key]

    const int tid  = threadIdx.x;
    const int lane = tid & 31;
    const int wave = tid >> 5;
    const int b     = blockIdx.y;
    const int qtile = blockIdx.x;
    const int qbase = qtile * 128 + wave * 16;

    const uint32_t kOff[2] = { (uint32_t)(uintptr_t)&kTile[0][0],
                               (uint32_t)(uintptr_t)&kTile[1][0] };
    const uint32_t vOff[2] = { (uint32_t)(uintptr_t)&vTile[0][0],
                               (uint32_t)(uintptr_t)&vTile[1][0] };

    // Q fragments (K-dim = h), kept in registers: 4 chunks of 16x32
    v16bf qa[4];
    {
        const __bf16* qrow = q + ((size_t)b * SEQ + qbase + (lane & 15)) * HDIM
                               + (lane >> 4) * 16;
#pragma unroll
        for (int c = 0; c < 4; ++c) qa[c] = load16bf(qrow + c * 32);
    }

    v8f o[8];
#pragma unroll
    for (int f = 0; f < 8; ++f) o[f] = zero8();
    float m[8], l[8];
#pragma unroll
    for (int r = 0; r < 8; ++r) { m[r] = -1e30f; l[r] = 0.0f; }

    const float scale = 0.08838834764831845f;   // 1/sqrt(128)
    const int   nkb   = qtile * 4 + 4;          // 32-key blocks incl. diagonal
    const int   qrow_off = (lane >> 4) * 8;     // C-layout row offset for this lane

    // Prologue: DMA key-block 0 into buffer 0.
    if (wave == 0) {
        const __bf16* ksrc = kmat + ((size_t)b * SEQ) * HDIM;
        const __bf16* vsrc = vT + (size_t)b * HDIM * SEQ;
        // K tile: 128 x 32 rows, row = 64 DWORDs -> pad codes (5,3) => 272B pitch
        tdm_load_2d(kOff[0], ksrc, HDIM, HDIM, 32, HDIM, 5u, 3u);
        // V tile: 32 x 128 rows, row = 16 DWORDs -> pad codes (3,3) => 80B pitch
        tdm_load_2d(vOff[0], vsrc, SEQ, 32, HDIM, SEQ, 3u, 3u);
        __builtin_amdgcn_s_wait_tensorcnt(0);
    }
    __syncthreads();

    for (int kb = 0; kb < nkb; ++kb) {
        const int cur = kb & 1;

        // Prefetch next key block into the other buffer (TDM, async).
        if (wave == 0 && (kb + 1) < nkb) {
            const int nb = (kb + 1) & 1;
            const __bf16* ksrc = kmat + ((size_t)b * SEQ + (kb + 1) * 32) * HDIM;
            const __bf16* vsrc = vT + (size_t)b * HDIM * SEQ + (kb + 1) * 32;
            tdm_load_2d(kOff[nb], ksrc, HDIM, HDIM, 32, HDIM, 5u, 3u);
            tdm_load_2d(vOff[nb], vsrc, SEQ, 32, HDIM, SEQ, 3u, 3u);
        }

        if (kb * 32 <= qbase + 15) {            // not fully masked for this wave
            const __bf16* kt = &kTile[cur][0];
            const __bf16* vt = &vTile[cur][0];

            // S = Q K^T, two 16-key subtiles, K-dim = h (4 x 32)
            v8f s[2];
#pragma unroll
            for (int ks = 0; ks < 2; ++ks) {
                v8f accs = zero8();
#pragma unroll
                for (int c = 0; c < 4; ++c) {
                    const v16bf bf = load16bf(
                        &kt[(ks * 16 + (lane & 15)) * PITCH_KV + c * 32 + (lane >> 4) * 16]);
                    accs = wmma_bf16(qa[c], bf, accs);
                }
                s[ks] = accs;
            }

            // Scale + causal mask (query = qbase+qrow_off+r, key via lane)
#pragma unroll
            for (int ks = 0; ks < 2; ++ks) {
                const int key = kb * 32 + ks * 16 + (lane & 15);
#pragma unroll
                for (int r = 0; r < 8; ++r) {
                    const float sv = s[ks][r] * scale;
                    s[ks][r] = (key <= qbase + qrow_off + r) ? sv : -1e30f;
                }
            }

            // Online softmax: rows live across 16-lane halves
#pragma unroll
            for (int r = 0; r < 8; ++r) {
                float rm = fmaxf(s[0][r], s[1][r]);
#pragma unroll
                for (int off = 8; off >= 1; off >>= 1)
                    rm = fmaxf(rm, __shfl_xor(rm, off, 32));
                const float mn = fmaxf(m[r], rm);
                const float sc = __expf(m[r] - mn);
                const float p0 = __expf(s[0][r] - mn);
                const float p1 = __expf(s[1][r] - mn);
                s[0][r] = p0; s[1][r] = p1;
                float rs = p0 + p1;
#pragma unroll
                for (int off = 8; off >= 1; off >>= 1)
                    rs += __shfl_xor(rs, off, 32);
                l[r] = l[r] * sc + rs;
                m[r] = mn;
#pragma unroll
                for (int f = 0; f < 8; ++f) o[f][r] *= sc;
            }

            // P: C-layout -> bf16 A-fragment via wave-private LDS bounce
            {
                __bf16* pw = pTile[wave];
#pragma unroll
                for (int ks = 0; ks < 2; ++ks)
#pragma unroll
                    for (int r = 0; r < 8; ++r)
                        pw[(qrow_off + r) * PITCH_P + ks * 16 + (lane & 15)] =
                            (__bf16)s[ks][r];
                const v16bf pa = load16bf(&pw[(lane & 15) * PITCH_P + (lane >> 4) * 16]);
                // O += P V : 8 h-subtiles, K-dim = 32 keys
#pragma unroll
                for (int f = 0; f < 8; ++f) {
                    const v16bf bf = load16bf(
                        &vt[(f * 16 + (lane & 15)) * PITCH_P + (lane >> 4) * 16]);
                    o[f] = wmma_bf16(pa, bf, o[f]);
                }
            }
        }

        // Ensure prefetch landed, then publish to all waves.
        if (wave == 0) __builtin_amdgcn_s_wait_tensorcnt(0);
        __syncthreads();
    }

    // Epilogue: divide by row sums, store fp32
    float invl[8];
#pragma unroll
    for (int r = 0; r < 8; ++r) invl[r] = 1.0f / l[r];
#pragma unroll
    for (int f = 0; f < 8; ++f) {
        const int h = f * 16 + (lane & 15);
#pragma unroll
        for (int r = 0; r < 8; ++r) {
            const int row = qbase + qrow_off + r;
            out[((size_t)b * SEQ + row) * HDIM + h] = o[f][r] * invl[r];
        }
    }
}

extern "C" void kernel_launch(void* const* d_in, const int* in_sizes, int n_in,
                              void* d_out, int out_size, void* d_ws, size_t ws_size,
                              hipStream_t stream) {
    (void)in_sizes; (void)n_in; (void)out_size; (void)ws_size;
    const float* x  = (const float*)d_in[0];
    const float* Wk = (const float*)d_in[1];
    const float* Wq = (const float*)d_in[2];
    const float* Wv = (const float*)d_in[3];
    float* out = (float*)d_out;

    const size_t QKV_BYTES = (size_t)BATCH * SEQ * HDIM * sizeof(__bf16); // 8 MB
    char* ws = (char*)d_ws;
    __bf16* kb  = (__bf16*)(ws);
    __bf16* qb  = (__bf16*)(ws + QKV_BYTES);
    __bf16* vtb = (__bf16*)(ws + 2 * QKV_BYTES);

    dim3 gproj((BATCH * SEQ) / 128, 1, 3);
    qkv_proj_kernel<<<gproj, 256, 0, stream>>>(x, Wk, Wq, Wv, kb, qb, vtb);

    dim3 gattn(SEQ / 128, BATCH);
    attn_kernel<<<gattn, 256, 0, stream>>>(qb, kb, vtb, out);
}
